// SelfAttention_41291815584099
// MI455X (gfx1250) — matrix operand order repaired and checked by reading, not compiled, not run on hardware
//
#include <hip/hip_runtime.h>

// ---------------------------------------------------------------------------
// Self-attention (SAGAN style) for MI455X / gfx1250, bf16 WMMA pipeline.
// B=8, H=W=64, C=512, C8=64, C2=256, N=4096, NK=1024 (pooled keys)
// ---------------------------------------------------------------------------

typedef __attribute__((ext_vector_type(16))) __bf16 v16bf;
typedef __attribute__((ext_vector_type(8)))  float  v8f;

union Frag {
    v16bf v;
    unsigned short u[16];
    uint4 q[2];
};

__device__ __forceinline__ unsigned short f2bf(float f) {
    unsigned u = __float_as_uint(f);
    unsigned r = u + 0x7FFFu + ((u >> 16) & 1u);   // round-to-nearest-even
    return (unsigned short)(r >> 16);
}
__device__ __forceinline__ float bf2f(unsigned short u) {
    return __uint_as_float(((unsigned)u) << 16);
}

__device__ __forceinline__ v8f wmma_bf16(const v16bf a, const v16bf b, v8f c) {
    // D = A(16x32 bf16) * B(32x16 bf16) + C(16x16 f32)
    return __builtin_amdgcn_wmma_f32_16x16x32_bf16(
        false, a, false, b, (short)0, c, false, false);
}

__device__ __forceinline__ float redmax16(float v) {
    v = fmaxf(v, __shfl_xor(v, 1, 32));
    v = fmaxf(v, __shfl_xor(v, 2, 32));
    v = fmaxf(v, __shfl_xor(v, 4, 32));
    v = fmaxf(v, __shfl_xor(v, 8, 32));
    return v;
}
__device__ __forceinline__ float redsum16(float v) {
    v += __shfl_xor(v, 1, 32);
    v += __shfl_xor(v, 2, 32);
    v += __shfl_xor(v, 4, 32);
    v += __shfl_xor(v, 8, 32);
    return v;
}

// ---------------------------------------------------------------------------
// Kernel 0: convert weights to bf16, stored [N][K] (transposed) so B-fragments
// are two contiguous 8-element runs per lane.
// Wt[384][512]: cols 0-63 = wf, 64-127 = wg, 128-383 = wh.  woT[512][256].
// ---------------------------------------------------------------------------
__global__ void k_wconv(const float* __restrict__ wf, const float* __restrict__ wg,
                        const float* __restrict__ wh, const float* __restrict__ wo,
                        unsigned short* __restrict__ Wt, unsigned short* __restrict__ woT) {
    int tid = blockIdx.x * blockDim.x + threadIdx.x;
    if (tid < 384 * 512) {
        int n = tid >> 9, k = tid & 511;
        float val = (n < 64)  ? wf[k * 64 + n]
                  : (n < 128) ? wg[k * 64 + (n - 64)]
                              : wh[k * 256 + (n - 128)];
        Wt[tid] = f2bf(val);
    }
    if (tid < 512 * 256) {
        int n = tid >> 8, k = tid & 255;
        woT[tid] = f2bf(wo[k * 512 + n]);  // wo is [256][512]
    }
}

// ---------------------------------------------------------------------------
// Kernel 1: y = x @ [wf|wg|wh]   (32768 x 512) @ (512 x 384)
// 512 blocks x 256 threads (8 waves). Wave (w): rows 16*(w>>1), cols 192*(w&1).
// ---------------------------------------------------------------------------
__global__ __launch_bounds__(256) void k_proj(const float* __restrict__ x,
                                              const unsigned short* __restrict__ Wt,
                                              unsigned short* __restrict__ yf,   // [32768][64]
                                              unsigned short* __restrict__ g,    // [32768][64]
                                              unsigned short* __restrict__ yh) { // [32768][256]
    const int lane = threadIdx.x & 31, w = threadIdx.x >> 5;
    const int r0 = blockIdx.x * 64 + (w >> 1) * 16;
    const int c0 = (w & 1) * 192;
    const int m = lane & 15, kh = lane >> 4;

    v8f acc[12] = {};
    for (int kk = 0; kk < 512; kk += 32) {
        // A fragment: 2 contiguous 8-float chunks per lane, K = kk+kh*8.. and kk+16+kh*8..
        Frag a;
        const float* xp = x + (size_t)(r0 + m) * 512 + kk + kh * 8;
        float4 f0 = *(const float4*)(xp);
        float4 f1 = *(const float4*)(xp + 4);
        float4 f2 = *(const float4*)(xp + 16);
        float4 f3 = *(const float4*)(xp + 20);
        a.u[0] = f2bf(f0.x); a.u[1] = f2bf(f0.y); a.u[2] = f2bf(f0.z); a.u[3] = f2bf(f0.w);
        a.u[4] = f2bf(f1.x); a.u[5] = f2bf(f1.y); a.u[6] = f2bf(f1.z); a.u[7] = f2bf(f1.w);
        a.u[8]  = f2bf(f2.x); a.u[9]  = f2bf(f2.y); a.u[10] = f2bf(f2.z); a.u[11] = f2bf(f2.w);
        a.u[12] = f2bf(f3.x); a.u[13] = f2bf(f3.y); a.u[14] = f2bf(f3.z); a.u[15] = f2bf(f3.w);

#pragma unroll
        for (int t = 0; t < 12; t++) {
            Frag b;
            const unsigned short* bp = Wt + (size_t)(c0 + t * 16 + m) * 512 + kk + kh * 8;
            b.q[0] = *(const uint4*)bp;
            b.q[1] = *(const uint4*)(bp + 16);
            acc[t] = wmma_bf16(a.v, b.v, acc[t]);
        }
    }
    // C layout: VGPR v -> row r0 + v + 8*kh, col = c0 + t*16 + m
#pragma unroll
    for (int t = 0; t < 12; t++) {
        int col = c0 + t * 16 + m;
#pragma unroll
        for (int v = 0; v < 8; v++) {
            int row = r0 + v + 8 * kh;
            unsigned short val = f2bf(acc[t][v]);
            if (col < 64)       yf[(size_t)row * 64 + col]          = val;
            else if (col < 128) g [(size_t)row * 64 + (col - 64)]   = val;
            else                yh[(size_t)row * 256 + (col - 128)] = val;
        }
    }
}

// ---------------------------------------------------------------------------
// Pooling: 2x2 valid max-pool.  f keeps [B][1024][64]; h is written transposed
// as hT[B][256][1024] so o-GEMM B-fragments are contiguous.
// ---------------------------------------------------------------------------
__global__ void k_poolf(const unsigned short* __restrict__ yf, unsigned short* __restrict__ f) {
    int tid = blockIdx.x * blockDim.x + threadIdx.x;
    if (tid >= 8 * 1024 * 64) return;
    int d = tid & 63, p = (tid >> 6) & 1023, b = tid >> 16;
    int ph = p >> 5, pw = p & 31;
    size_t base = ((size_t)b * 4096 + ph * 128 + pw * 2) * 64 + d;
    float m0 = fmaxf(bf2f(yf[base]),        bf2f(yf[base + 64]));
    float m1 = fmaxf(bf2f(yf[base + 4096]), bf2f(yf[base + 4160]));
    f[tid] = f2bf(fmaxf(m0, m1));
}

__global__ void k_poolh(const unsigned short* __restrict__ yh, unsigned short* __restrict__ hT) {
    int tid = blockIdx.x * blockDim.x + threadIdx.x;
    if (tid >= 8 * 256 * 1024) return;
    int p = tid & 1023, d = (tid >> 10) & 255, b = tid >> 18;
    int ph = p >> 5, pw = p & 31;
    size_t base = ((size_t)b * 4096 + ph * 128 + pw * 2) * 256 + d;
    float m0 = fmaxf(bf2f(yh[base]),         bf2f(yh[base + 256]));
    float m1 = fmaxf(bf2f(yh[base + 16384]), bf2f(yh[base + 16640]));
    hT[tid] = f2bf(fmaxf(m0, m1));  // hT[b][d][p]
}

// ---------------------------------------------------------------------------
// Kernel 2: streaming-softmax attention.
// 512 blocks (8 batches x 64 query groups) x 128 threads (4 waves).
// Wave: 16 queries, o accum 16x256 f32 (16 WMMA tiles), 16 key-tiles of 64.
// ---------------------------------------------------------------------------
__global__ __launch_bounds__(128) void k_attn(const unsigned short* __restrict__ g,   // [32768][64]
                                              const unsigned short* __restrict__ f,   // [B][1024][64]
                                              const unsigned short* __restrict__ hT,  // [B][256][1024]
                                              unsigned short* __restrict__ o) {       // [32768][256]
    __shared__ unsigned short pbuf[4][16][80];  // per-wave p tile, 160B row stride (16B aligned)
    const int lane = threadIdx.x & 31, w = threadIdx.x >> 5;
    const int b = blockIdx.x >> 6, qg = blockIdx.x & 63;
    const int row0 = b * 4096 + qg * 64 + w * 16;
    const int m = lane & 15, kh = lane >> 4;

    // Query A-fragments (K = channel dim 64 -> 2 fragments), loaded once.
    Frag ag[2];
#pragma unroll
    for (int ks = 0; ks < 2; ks++) {
        const unsigned short* gp = g + (size_t)(row0 + m) * 64 + ks * 32 + kh * 8;
        ag[ks].q[0] = *(const uint4*)gp;
        ag[ks].q[1] = *(const uint4*)(gp + 16);
    }

    float mrun[8], lrun[8];
#pragma unroll
    for (int v = 0; v < 8; v++) { mrun[v] = -1e30f; lrun[v] = 0.f; }
    v8f acc[16] = {};

    for (int kt = 0; kt < 16; kt++) {
        // ---- scores s[16q x 64k] ----
        v8f s[4] = {};
#pragma unroll
        for (int ks = 0; ks < 2; ks++) {
#pragma unroll
            for (int t = 0; t < 4; t++) {
                Frag bf_;
                const unsigned short* fp =
                    f + ((size_t)b * 1024 + kt * 64 + t * 16 + m) * 64 + ks * 32 + kh * 8;
                bf_.q[0] = *(const uint4*)fp;
                bf_.q[1] = *(const uint4*)(fp + 16);
                s[t] = wmma_bf16(ag[ks].v, bf_.v, s[t]);
            }
        }
        // ---- online softmax (row = v + 8*kh, 16 cols per lane-group) ----
        float scale[8];
#pragma unroll
        for (int v = 0; v < 8; v++) {
            float tm = fmaxf(fmaxf(s[0][v], s[1][v]), fmaxf(s[2][v], s[3][v]));
            tm = redmax16(tm);
            float mn = fmaxf(mrun[v], tm);
            scale[v] = __expf(mrun[v] - mn);
            float rs = 0.f;
#pragma unroll
            for (int t = 0; t < 4; t++) {
                float p = __expf(s[t][v] - mn);
                s[t][v] = p;
                rs += p;
            }
            rs = redsum16(rs);
            lrun[v] = lrun[v] * scale[v] + rs;
            mrun[v] = mn;
        }
#pragma unroll
        for (int nt = 0; nt < 16; nt++)
#pragma unroll
            for (int v = 0; v < 8; v++) acc[nt][v] *= scale[v];

        // ---- C-layout -> A-layout via wave-private LDS round trip ----
#pragma unroll
        for (int t = 0; t < 4; t++)
#pragma unroll
            for (int v = 0; v < 8; v++)
                pbuf[w][v + 8 * kh][t * 16 + m] = f2bf(s[t][v]);

        Frag ap[2];
#pragma unroll
        for (int ks = 0; ks < 2; ks++) {
            const unsigned short* pp = &pbuf[w][m][ks * 32 + kh * 8];
            ap[ks].q[0] = *(const uint4*)pp;
            ap[ks].q[1] = *(const uint4*)(pp + 16);
        }

        // ---- o += p @ h_tile ----
#pragma unroll
        for (int nt = 0; nt < 16; nt++) {
#pragma unroll
            for (int ks = 0; ks < 2; ks++) {
                Frag bh;
                const unsigned short* hp =
                    hT + ((size_t)b * 256 + nt * 16 + m) * 1024 + kt * 64 + ks * 32 + kh * 8;
                bh.q[0] = *(const uint4*)hp;
                bh.q[1] = *(const uint4*)(hp + 16);
                acc[nt] = wmma_bf16(ap[ks].v, bh.v, acc[nt]);
            }
        }
    }
    // ---- normalize + store o bf16 ----
#pragma unroll
    for (int nt = 0; nt < 16; nt++)
#pragma unroll
        for (int v = 0; v < 8; v++) {
            int row = row0 + v + 8 * kh;
            o[(size_t)row * 256 + nt * 16 + m] = f2bf(acc[nt][v] / lrun[v]);
        }
}

// ---------------------------------------------------------------------------
// Kernel 3: out = gamma * (o @ wo) + x.  (32768 x 256) @ (256 x 512)
// 512 blocks x 256 threads (8 waves): wave = 16 rows x 256 cols.
// ---------------------------------------------------------------------------
__global__ __launch_bounds__(256) void k_out(const unsigned short* __restrict__ o,
                                             const unsigned short* __restrict__ woT, // [512][256]
                                             const float* __restrict__ x,
                                             const float* __restrict__ gamma,
                                             float* __restrict__ out) {
    const int lane = threadIdx.x & 31, w = threadIdx.x >> 5;
    const int r0 = blockIdx.x * 64 + (w >> 1) * 16;
    const int c0 = (w & 1) * 256;
    const int m = lane & 15, kh = lane >> 4;

    v8f acc[16] = {};
    for (int kk = 0; kk < 256; kk += 32) {
        Frag a;
        const unsigned short* ap = o + (size_t)(r0 + m) * 256 + kk + kh * 8;
        a.q[0] = *(const uint4*)ap;
        a.q[1] = *(const uint4*)(ap + 16);
#pragma unroll
        for (int t = 0; t < 16; t++) {
            Frag b;
            const unsigned short* bp = woT + (size_t)(c0 + t * 16 + m) * 256 + kk + kh * 8;
            b.q[0] = *(const uint4*)bp;
            b.q[1] = *(const uint4*)(bp + 16);
            acc[t] = wmma_bf16(a.v, b.v, acc[t]);
        }
    }
    const float gm = gamma[0];
#pragma unroll
    for (int t = 0; t < 16; t++) {
        int col = c0 + t * 16 + m;
#pragma unroll
        for (int v = 0; v < 8; v++) {
            size_t idx = (size_t)(r0 + v + 8 * kh) * 512 + col;
            out[idx] = gm * acc[t][v] + x[idx];
        }
    }
}

// ---------------------------------------------------------------------------
// Launch
// ---------------------------------------------------------------------------
extern "C" void kernel_launch(void* const* d_in, const int* in_sizes, int n_in,
                              void* d_out, int out_size, void* d_ws, size_t ws_size,
                              hipStream_t stream) {
    const float* x     = (const float*)d_in[0];
    const float* wf    = (const float*)d_in[1];
    const float* wg    = (const float*)d_in[2];
    const float* wh    = (const float*)d_in[3];
    const float* wo    = (const float*)d_in[4];
    const float* gamma = (const float*)d_in[5];
    float* out = (float*)d_out;

    char* ws = (char*)d_ws;
    size_t off = 0;
    auto alloc = [&](size_t bytes) { char* p = ws + off; off += (bytes + 255) & ~(size_t)255; return p; };
    unsigned short* yf  = (unsigned short*)alloc((size_t)32768 * 64 * 2);   // pre-pool f
    unsigned short* yh  = (unsigned short*)alloc((size_t)32768 * 256 * 2);  // pre-pool h
    unsigned short* g   = (unsigned short*)alloc((size_t)32768 * 64 * 2);
    unsigned short* f   = (unsigned short*)alloc((size_t)8 * 1024 * 64 * 2);
    unsigned short* hT  = (unsigned short*)alloc((size_t)8 * 256 * 1024 * 2);
    unsigned short* ob  = (unsigned short*)alloc((size_t)32768 * 256 * 2);
    unsigned short* Wt  = (unsigned short*)alloc((size_t)384 * 512 * 2);
    unsigned short* woT = (unsigned short*)alloc((size_t)512 * 256 * 2);

    k_wconv<<<(384 * 512 + 255) / 256, 256, 0, stream>>>(wf, wg, wh, wo, Wt, woT);
    k_proj<<<512, 256, 0, stream>>>(x, Wt, yf, g, yh);
    k_poolf<<<(8 * 1024 * 64) / 256, 256, 0, stream>>>(yf, f);
    k_poolh<<<(8 * 256 * 1024) / 256, 256, 0, stream>>>(yh, hT);
    k_attn<<<512, 128, 0, stream>>>(g, f, hT, ob);
    k_out<<<512, 256, 0, stream>>>(ob, woT, x, gamma, out);
}
